// HPCrnn_67542655696941
// MI455X (gfx1250) — matrix operand order repaired and checked by reading, not compiled, not run on hardware
//
#include <hip/hip_runtime.h>
#include <hip/hip_bf16.h>
#include <stdint.h>

// ---------------------------------------------------------------------------
// HPC RNN scan for MI455X (gfx1250, wave32, WMMA).
//   bs=256, ecnum=ca1num=ca3num=1024, T=100, actnum=2
// Batch rows evolve independently -> 16 persistent workgroups, each owning a
// 16-row group. Per step: two 16x1024x1024 GEMMs via v_wmma_f32_16x16x32_bf16
// (fp32 accumulate). State lives in registers in WMMA C/D layout; the shared
// A operand is staged in two XOR-swizzled 32KB LDS buffers (2 barriers/step).
// Weights are bf16, pre-transposed, streamed from L2 every step. A laundered
// integer offset (not a laundered pointer!) keeps the loads loop-variant so
// LICM can't hoist+spill them, while preserving addrspace(1) provenance so
// they lower to global_load_b128 (saddr form), not flat_load_b128.
// ---------------------------------------------------------------------------

#define ECN 1024
#define BS  256
#define TT  100
#define CUE_T 10
#define NT  4                   // N-tiles (of 16 cols) per wave

typedef __attribute__((ext_vector_type(16))) __bf16       v16bf;
typedef __attribute__((ext_vector_type(8)))  float        v8f;
typedef __attribute__((ext_vector_type(4)))  unsigned int u32x4;

struct Frag32B { u32x4 lo, hi; };

// Build a 16x bf16 fragment from two 16-byte chunks (ds_load_b128 / global_load_b128)
__device__ __forceinline__ v16bf make_frag(const __bf16* p0, const __bf16* p1) {
  Frag32B f;
  f.lo = *(const u32x4*)p0;
  f.hi = *(const u32x4*)p1;
  return __builtin_bit_cast(v16bf, f);
}

__device__ __forceinline__ float sigmoidf_fast(float x) {
  return 1.0f / (1.0f + __expf(-x));
}

// XOR bank swizzle: rotate 16B chunks within each 128B window by row (m).
// Element index for (row m, element e):  m*1024 + (e ^ ((m&7)<<3)).
__device__ __forceinline__ int sw_idx(int m, int e) {
  return m * 1024 + (e ^ ((m & 7) << 3));
}

// ---------------------------------------------------------------------------
// Kernel 1: fp32 -> bf16 transpose (W[k][n] -> Wt[n][k]) via LDS tile.
// ---------------------------------------------------------------------------
__global__ __launch_bounds__(1024)
void hpc_tr_kernel(const float* __restrict__ W, __bf16* __restrict__ Wt) {
  __shared__ float tile[32][33];
  const int k0 = blockIdx.x * 32, n0 = blockIdx.y * 32;
  const int tx = threadIdx.x, ty = threadIdx.y;
  tile[ty][tx] = W[(k0 + ty) * ECN + (n0 + tx)];
  __syncthreads();
  Wt[(n0 + ty) * ECN + (k0 + tx)] = (__bf16)tile[tx][ty];
}

// ---------------------------------------------------------------------------
// Kernel 2: ca3proj[t][j] = sum_i exp(-(c_i - t)^2 / 50) * wca3ca1[i][j]
// ---------------------------------------------------------------------------
__global__ __launch_bounds__(256)
void hpc_proj_kernel(const float* __restrict__ wca3ca1, float* __restrict__ proj) {
  __shared__ float gauss[ECN];
  const int t = blockIdx.x;
  const int j = blockIdx.y * 256 + threadIdx.x;
  #pragma unroll
  for (int r = 0; r < 4; ++r) {
    const int i = threadIdx.x + r * 256;
    const float c = -10.0f + 120.0f * (float)i * (1.0f / 1023.0f);
    const float d = c - (float)t;
    gauss[i] = __expf(-d * d * (1.0f / 50.0f));
  }
  __syncthreads();
  float s = 0.0f;
  for (int i = 0; i < ECN; ++i) s = fmaf(gauss[i], wca3ca1[i * ECN + j], s);
  proj[t * ECN + j] = s;
}

// ---------------------------------------------------------------------------
// Kernel 3: persistent scan. 16 blocks (one per 16-row batch group),
// 512 threads = 16 waves, each wave owns 4 N-tiles of 16 columns.
// ---------------------------------------------------------------------------
__global__ __launch_bounds__(512)
void hpc_scan_kernel(const int*   __restrict__ cue,
                     const float* __restrict__ ec3_last,
                     const float* __restrict__ ec5_last,
                     const float* __restrict__ noise,    // (T, BS, ECN)
                     const float* __restrict__ ca1bias,
                     const __bf16* __restrict__ W1t,     // wec3ca1^T  (n-major)
                     const __bf16* __restrict__ W2t,     // wca1ec5^T  (n-major)
                     const float* __restrict__ proj,     // (T, ECN)
                     float* __restrict__ out_ec3his,
                     float* __restrict__ out_ec5his,
                     float* __restrict__ out_ca1his,
                     float* __restrict__ out_ec3,
                     float* __restrict__ out_ec5,
                     float* __restrict__ out_ca1)
{
  // Ab0: ec3_expose operand, Ab1: ca1 operand (fixed roles; XOR-swizzled rows)
  __shared__ __align__(16) __bf16 Ab0[16 * 1024];
  __shared__ __align__(16) __bf16 Ab1[16 * 1024];

  const int tid   = threadIdx.x;
  const int wave  = tid >> 5;        // 0..15
  const int lane  = tid & 31;
  const int hi    = lane >> 4;       // K-half selector for fragments
  const int nloc  = lane & 15;       // A: M row index; B: N column index
  const int group = blockIdx.x;      // 0..15 batch group
  const int nbase = wave * (16 * NT);
  const int xr    = (nloc & 7) << 3; // per-lane XOR swizzle for A-fragment reads

  v8f ec3[NT], ec5[NT], ca1[NT];

  // laundered zero: keeps weight-load addresses loop-variant across t while
  // preserving the kernel-arg pointer provenance (=> global_load, not flat)
  int lz = 0;

  // ---- init: load state in WMMA C/D layout, stage ec3_expose(0)=ec3_last ----
  #pragma unroll
  for (int ct = 0; ct < NT; ++ct) {
    const int col = nbase + ct * 16 + nloc;
    #pragma unroll
    for (int v = 0; v < 8; ++v) {
      const int m   = v + hi * 8;                // C/D layout row
      const int row = group * 16 + m;            // global batch row
      const float e3 = ec3_last[row * ECN + col];
      ec3[ct][v] = e3;
      ec5[ct][v] = ec5_last[row * ECN + col];
      ca1[ct][v] = 0.0f;
      Ab0[sw_idx(m, col)] = (__bf16)e3;
    }
  }
  __syncthreads();

  for (int t = 0; t < TT; ++t) {
    asm volatile("" : "+s"(lz));                 // opaque 0, redefined per step
    const __bf16* __restrict__ W1p = W1t + lz;   // still addrspace(1)-rooted
    const __bf16* __restrict__ W2p = W2t + lz;

    // -- prefetch this step's pink-noise block (streaming; NT hint, skips L2 churn)
    v8f nz[NT];
    #pragma unroll
    for (int ct = 0; ct < NT; ++ct) {
      const int col = nbase + ct * 16 + nloc;
      #pragma unroll
      for (int v = 0; v < 8; ++v) {
        const int row = group * 16 + v + hi * 8;
        nz[ct][v] = __builtin_nontemporal_load(
            &noise[(size_t)(t * BS + row) * ECN + col]);
      }
    }

    // ================= GEMM1: raw = ec3_expose @ W1 (A: Ab0, B: L2) ==========
    v8f acc[NT] = {};
    {
      const __bf16* arow = &Ab0[nloc * 1024];
      #pragma unroll 2
      for (int kb = 0; kb < ECN; kb += 32) {
        // A fragment: lane=M row, K halves {0-7,16-23} / {8-15,24-31}, swizzled
        const v16bf af = make_frag(arow + ((kb + hi * 8) ^ xr),
                                   arow + ((kb + 16 + hi * 8) ^ xr));
        #pragma unroll
        for (int ct = 0; ct < NT; ++ct) {
          // B fragment: lane=N col of transposed weight, K halves {0-7,16-23} / {8-15,24-31}
          const __bf16* bp = W1p + (size_t)(nbase + ct * 16 + nloc) * ECN + kb + hi * 8;
          const v16bf bf = make_frag(bp, bp + 16);
          acc[ct] = __builtin_amdgcn_wmma_f32_16x16x32_bf16(
              false, af, false, bf, (short)0, acc[ct], false, false);
        }
      }
    }

    // ca1 = relu(proj * (1 + sigmoid(raw)) - bias); stage bf16 into Ab1
    #pragma unroll
    for (int ct = 0; ct < NT; ++ct) {
      const int col = nbase + ct * 16 + nloc;
      const float p = proj[t * ECN + col];
      const float b = ca1bias[col];
      #pragma unroll
      for (int v = 0; v < 8; ++v) {
        const float c = p * (1.0f + sigmoidf_fast(acc[ct][v])) - b;
        const float r = c > 0.0f ? c : 0.0f;
        ca1[ct][v] = r;
        Ab1[sw_idx(v + hi * 8, col)] = (__bf16)r;
      }
    }
    __syncthreads();   // Ab1 complete; also: all waves are done reading Ab0

    // ================= GEMM2: ec5 += ca1 @ W2  (C = ec5, 10*TS == 1.0) =======
    {
      const __bf16* arow = &Ab1[nloc * 1024];
      #pragma unroll 2
      for (int kb = 0; kb < ECN; kb += 32) {
        const v16bf af = make_frag(arow + ((kb + hi * 8) ^ xr),
                                   arow + ((kb + 16 + hi * 8) ^ xr));
        #pragma unroll
        for (int ct = 0; ct < NT; ++ct) {
          const __bf16* bp = W2p + (size_t)(nbase + ct * 16 + nloc) * ECN + kb + hi * 8;
          const v16bf bf = make_frag(bp, bp + 16);
          ec5[ct] = __builtin_amdgcn_wmma_f32_16x16x32_bf16(
              false, af, false, bf, (short)0, ec5[ct], false, false);
        }
      }
    }

    // ====== elementwise: squash ec5, ec3 *= ec5, ec3_expose = ec3+noise ======
    // (writes Ab0 for step t+1; safe: every wave passed the mid barrier, i.e.
    //  finished its GEMM1 reads of Ab0)
    #pragma unroll
    for (int ct = 0; ct < NT; ++ct) {
      const int col = nbase + ct * 16 + nloc;
      float h_e3e = 0.0f, h_e5 = 0.0f;
      #pragma unroll
      for (int v = 0; v < 8; ++v) {
        const int m   = v + hi * 8;
        const int row = group * 16 + m;
        const float e5 = 0.7f + 0.3f * sigmoidf_fast(4.0f * (ec5[ct][v] - 0.3f));
        ec5[ct][v] = e5;
        float e3 = e5 * ec3[ct][v];
        const float e3e = e3 + nz[ct][v];
        Ab0[sw_idx(m, col)] = (__bf16)e3e;           // A operand for step t+1
        if (t == CUE_T) {                             // uniform branch
          const int cv = cue[row * ECN + col];
          e3 = cv ? fmaf(0.4f, e3, 0.6f) : e3;        // branchless per-lane select
        }
        ec3[ct][v] = e3;
        if (v == 0) { h_e3e = e3e; h_e5 = e5; }       // batch row 0 history
      }
      if (group == 0 && hi == 0) {                    // row m=0: v=0, lanes 0-15
        __builtin_nontemporal_store(h_e3e,      &out_ec3his[t * ECN + col]);
        __builtin_nontemporal_store(h_e5,       &out_ec5his[t * ECN + col]);
        __builtin_nontemporal_store(ca1[ct][0], &out_ca1his[t * ECN + col]);
      }
    }
    __syncthreads();   // Ab0(t+1) staged; all waves finished GEMM2 reads of Ab1
  }

  // ---- final carry states ----
  #pragma unroll
  for (int ct = 0; ct < NT; ++ct) {
    const int col = nbase + ct * 16 + nloc;
    #pragma unroll
    for (int v = 0; v < 8; ++v) {
      const int row = group * 16 + (v + hi * 8);
      __builtin_nontemporal_store(ec3[ct][v], &out_ec3[row * ECN + col]);
      __builtin_nontemporal_store(ec5[ct][v], &out_ec5[row * ECN + col]);
      __builtin_nontemporal_store(ca1[ct][v], &out_ca1[row * ECN + col]);
    }
  }
}

// ---------------------------------------------------------------------------
// Kernel 4: actCell = ca1 @ wca1act + actbias   (256x1024 @ 1024x2)
// ---------------------------------------------------------------------------
__global__ __launch_bounds__(512)
void hpc_act_kernel(const float* __restrict__ ca1,
                    const float* __restrict__ wca1act,
                    const float* __restrict__ actbias,
                    float* __restrict__ out_act) {
  const int tid = threadIdx.x;           // 512 = 256 rows x 2 outputs
  const int b = tid >> 1, a = tid & 1;
  float s = actbias[a];
  for (int j = 0; j < ECN; ++j) s = fmaf(ca1[b * ECN + j], wca1act[j * 2 + a], s);
  out_act[b * 2 + a] = s;
}

// ---------------------------------------------------------------------------
extern "C" void kernel_launch(void* const* d_in, const int* in_sizes, int n_in,
                              void* d_out, int out_size, void* d_ws, size_t ws_size,
                              hipStream_t stream) {
  const int*   cue      = (const int*)  d_in[0];
  const float* ec3_last = (const float*)d_in[1];
  const float* ec5_last = (const float*)d_in[2];
  // d_in[3] = ca1_last: carried but never read by the scan body -> unused
  const float* noise    = (const float*)d_in[4];
  const float* ca1bias  = (const float*)d_in[5];
  const float* wca3ca1  = (const float*)d_in[6];
  const float* wec3ca1  = (const float*)d_in[7];
  const float* wca1ec5  = (const float*)d_in[8];
  const float* wca1act  = (const float*)d_in[9];
  const float* actbias  = (const float*)d_in[10];

  // workspace: bf16 transposed weights (2 x 2MB) + fp32 ca3proj (400KB)
  __bf16* W1t  = (__bf16*)d_ws;
  __bf16* W2t  = W1t + (size_t)ECN * ECN;
  float*  proj = (float*)(W2t + (size_t)ECN * ECN);

  // outputs, concatenated flat in reference return order
  float* out        = (float*)d_out;
  float* out_act    = out;                       // (256, 2)
  float* out_ec3his = out + BS * 2;              // (100, 1024)
  float* out_ec5his = out_ec3his + TT * ECN;
  float* out_ca1his = out_ec5his + TT * ECN;
  float* out_ec3    = out_ca1his + TT * ECN;     // (256, 1024)
  float* out_ec5    = out_ec3 + BS * ECN;
  float* out_ca1    = out_ec5 + BS * ECN;

  hpc_tr_kernel<<<dim3(32, 32), dim3(32, 32), 0, stream>>>(wec3ca1, W1t);
  hpc_tr_kernel<<<dim3(32, 32), dim3(32, 32), 0, stream>>>(wca1ec5, W2t);
  hpc_proj_kernel<<<dim3(TT, 4), 256, 0, stream>>>(wca3ca1, proj);
  hpc_scan_kernel<<<16, 512, 0, stream>>>(cue, ec3_last, ec5_last, noise, ca1bias,
                                          W1t, W2t, proj,
                                          out_ec3his, out_ec5his, out_ca1his,
                                          out_ec3, out_ec5, out_ca1);
  hpc_act_kernel<<<1, 512, 0, stream>>>(out_ca1, wca1act, actbias, out_act);
}
